// PCooStep_23338852287251
// MI455X (gfx1250) — compile-verified, matrix operand order checked by reading, not run on hardware
//
#include <hip/hip_runtime.h>
#include <hip/hip_bf16.h>
#include <math.h>
#include <stdint.h>

typedef __attribute__((ext_vector_type(16))) _Float16 v16h;
typedef __attribute__((ext_vector_type(8)))  _Float16 v8h;
typedef __attribute__((ext_vector_type(8)))  float    v8f;

#define WPB 8  // waves per block (256 threads, wave32)

// ---------------- helpers: order-preserving float<->uint for atomicMax ----------
__device__ __forceinline__ unsigned f2key(float f) {
  unsigned b = __float_as_uint(f);
  return (b & 0x80000000u) ? ~b : (b | 0x80000000u);
}
__device__ __forceinline__ float key2f(unsigned k) {
  unsigned b = (k & 0x80000000u) ? (k ^ 0x80000000u) : ~k;
  return __uint_as_float(b);
}

// ---------------- prep kernels ----------------
__global__ void k_f32_to_f16(const float* __restrict__ x, _Float16* __restrict__ xh, int n) {
  int i = blockIdx.x * blockDim.x + threadIdx.x;
  if (i < n) xh[i] = (_Float16)x[i];
}

// Pre-pack the fused weight matrix [64 x NC] directly into WMMA B-fragment order:
//   Wfrag[(tn*2 + kb)*512 + lane*16 + j]  (f16)
// per ISA 7.12.2 (lane = n + 16g):
//   j in [0..7]  -> K = 32*kb + 8g + j
//   j in [8..15] -> K = 32*kb + 16 + 8g + (j-8)
// Column col = tn*16 + n selects one of 4 weight groups (q|k|v|s), each of
// logical width NC/4 with `dout` live columns (dout=64 for layers 1/2, 3 for layer 3).
__global__ void k_prep_frag(const float* __restrict__ Wq, const float* __restrict__ Wk,
                            const float* __restrict__ Wv, const float* __restrict__ Ws,
                            _Float16* __restrict__ Wfrag, int NC, int dout) {
  int i = blockIdx.x * blockDim.x + threadIdx.x;  // (NC/16)*1024 elements
  int total = (NC >> 4) * 1024;
  if (i >= total) return;
  int tn   = i >> 10;
  int rem  = i & 1023;
  int kb   = rem >> 9;
  int lane = (rem >> 4) & 31;
  int j    = rem & 15;
  int n = lane & 15, g = lane >> 4;
  int K = 32 * kb + ((j < 8) ? (8 * g + j) : (16 + 8 * g + (j - 8)));
  int col = tn * 16 + n;
  int gw  = NC >> 2;                 // group width
  int grp = col / gw, c = col % gw;
  const float* W = grp == 0 ? Wq : grp == 1 ? Wk : grp == 2 ? Wv : Ws;
  Wfrag[i] = (_Float16)((c < dout) ? W[K * dout + c] : 0.0f);
}

__global__ void k_prep_bias(const float* __restrict__ bq, const float* __restrict__ bk,
                            const float* __restrict__ bv, const float* __restrict__ bs,
                            float* __restrict__ bf, int NC, int dout) {
  int col = blockIdx.x * blockDim.x + threadIdx.x;
  if (col >= NC) return;
  int gw = NC >> 2;
  int grp = col / gw, c = col % gw;
  const float* b = grp == 0 ? bq : grp == 1 ? bk : grp == 2 ? bv : bs;
  bf[col] = (c < dout) ? b[c] : 0.0f;
}

// Reset segmented-softmax accumulators each call (graph-replay safe).
__global__ void k_init_seg(unsigned* __restrict__ mkey, float* __restrict__ ssum,
                           float* __restrict__ agg, int N, int aggTot) {
  int i = blockIdx.x * blockDim.x + threadIdx.x;
  if (i < N) {
    mkey[i] = f2key(-INFINITY);
    ssum[i] = 0.0f;
  }
  if (i < aggTot) agg[i] = 0.0f;
}

// ---------------- WMMA GEMM: Out[M x NC] = f16(X[M x 64]) @ f16(W[64 x NC]) + b --
// One wave computes a 16x64 output strip (4 N-tiles) from one A-tile load:
// 8 x v_wmma_f32_16x16x32_f16 per wave, B fragments pre-packed -> pure b128 loads.
__global__ void k_gemm_wmma(const _Float16* __restrict__ Xh,
                            const _Float16* __restrict__ Wfrag,
                            const float* __restrict__ bias,
                            float* __restrict__ Out,
                            int tiles_m, int NC) {
  int wave = threadIdx.x >> 5;
  int lane = threadIdx.x & 31;
  int ngrp_n = NC >> 6;  // groups of 4 N-tiles
  int group = blockIdx.x * WPB + wave;
  if (group >= tiles_m * ngrp_n) return;  // wave-uniform: EXEC stays all-1 for WMMA
  int tm  = group % tiles_m;              // consecutive waves share B fragments
  int tng = group / tiles_m;
  int l = lane & 15, g = lane >> 4;

  size_t row = (size_t)(tm * 16 + l);
  const v8h* xv = (const v8h*)(Xh + row * 64);  // 16B-aligned vector loads
  v8h A0 = xv[g], A1 = xv[2 + g], A2 = xv[4 + g], A3 = xv[6 + g];
  v16h a0 = __builtin_shufflevector(A0, A1, 0,1,2,3,4,5,6,7,8,9,10,11,12,13,14,15);
  v16h a1 = __builtin_shufflevector(A2, A3, 0,1,2,3,4,5,6,7,8,9,10,11,12,13,14,15);

  v8f acc[4] = {};
#pragma unroll
  for (int t = 0; t < 4; ++t) {
    int tn = tng * 4 + t;
    const _Float16* f0 = Wfrag + ((size_t)(tn * 2 + 0) << 9) + lane * 16;
    const _Float16* f1 = Wfrag + ((size_t)(tn * 2 + 1) << 9) + lane * 16;
    v8h B0a = *(const v8h*)f0, B0b = *(const v8h*)(f0 + 8);
    v8h B1a = *(const v8h*)f1, B1b = *(const v8h*)(f1 + 8);
    v16h b0 = __builtin_shufflevector(B0a, B0b, 0,1,2,3,4,5,6,7,8,9,10,11,12,13,14,15);
    v16h b1 = __builtin_shufflevector(B1a, B1b, 0,1,2,3,4,5,6,7,8,9,10,11,12,13,14,15);
    acc[t] = __builtin_amdgcn_wmma_f32_16x16x32_f16(false, a0, false, b0, (short)0, acc[t], false, false);
    acc[t] = __builtin_amdgcn_wmma_f32_16x16x32_f16(false, a1, false, b1, (short)0, acc[t], false, false);
  }

#pragma unroll
  for (int t = 0; t < 4; ++t) {
    int col = (tng * 4 + t) * 16 + l;
    float bb = bias[col];
#pragma unroll
    for (int r = 0; r < 8; ++r) {
      size_t m = (size_t)(tm * 16 + r + 8 * g);
      Out[m * NC + col] = acc[t][r] + bb;
    }
  }
}

// ---------------- edge kernels, D=64 (one wave per edge) ----------------
__global__ void k_edge_scores64(const float* __restrict__ QKVS, const int* __restrict__ src,
                                const int* __restrict__ dst, float* __restrict__ scores,
                                unsigned* __restrict__ mkey, int E, float scale) {
  int e = (int)((blockIdx.x * blockDim.x + threadIdx.x) >> 5);
  int lane = threadIdx.x & 31;
  if (e >= E) return;
  int s = src[e], d = dst[e];
  const float* q = QKVS + (size_t)d * 256;        // q at col 0
  const float* k = QKVS + (size_t)s * 256 + 64;   // k at col 64
  float p = q[lane] * k[lane] + q[lane + 32] * k[lane + 32];
#pragma unroll
  for (int off = 16; off; off >>= 1) p += __shfl_xor(p, off, 32);
  if (lane == 0) {
    float sc = p * scale;
    scores[e] = sc;
    atomicMax(mkey + d, f2key(sc));
  }
}

__global__ void k_edge_accum64(const float* __restrict__ QKVS, const float* __restrict__ scores,
                               const int* __restrict__ src, const int* __restrict__ dst,
                               const unsigned* __restrict__ mkey, float* __restrict__ ssum,
                               float* __restrict__ agg, int E) {
  int e = (int)((blockIdx.x * blockDim.x + threadIdx.x) >> 5);
  int lane = threadIdx.x & 31;
  if (e >= E) return;
  int s = src[e], d = dst[e];
  float w = __expf(scores[e] - key2f(mkey[d]));
  if (lane == 0) atomicAdd(ssum + d, w);
  const float* v = QKVS + (size_t)s * 256 + 128;  // v at col 128
  float* a = agg + (size_t)d * 64;
  atomicAdd(a + lane, v[lane] * w);
  atomicAdd(a + lane + 32, v[lane + 32] * w);
}

// normalize + skip (+residual) + tanh; emit f16 copy for next GEMM
__global__ void k_finalize64(const float* __restrict__ QKVS, const float* __restrict__ agg,
                             const float* __restrict__ ssum, float* __restrict__ h,
                             _Float16* __restrict__ hb, int N, int mode) {
  int i = blockIdx.x * blockDim.x + threadIdx.x;
  if (i >= N * 64) return;
  int n = i >> 6, c = i & 63;
  float out = agg[i] / (ssum[n] + 1e-16f) + QKVS[(size_t)n * 256 + 192 + c];  // skip at col 192
  float r = (mode == 2) ? tanhf(out + h[i]) : tanhf(out);
  h[i] = r;
  hb[i] = (_Float16)r;
}

// ---------------- edge kernels, D=3 (one thread per edge) ----------------
__global__ void k_edge3_scores(const float* __restrict__ O3, const int* __restrict__ src,
                               const int* __restrict__ dst, float* __restrict__ scores,
                               unsigned* __restrict__ mkey, int E, float scale) {
  int e = blockIdx.x * blockDim.x + threadIdx.x;
  if (e >= E) return;
  int s = src[e], d = dst[e];
  const float* q = O3 + (size_t)d * 64;        // q at col 0
  const float* k = O3 + (size_t)s * 64 + 16;   // k at col 16
  float sc = (q[0] * k[0] + q[1] * k[1] + q[2] * k[2]) * scale;
  scores[e] = sc;
  atomicMax(mkey + d, f2key(sc));
}

__global__ void k_edge3_accum(const float* __restrict__ O3, const float* __restrict__ scores,
                              const int* __restrict__ src, const int* __restrict__ dst,
                              const unsigned* __restrict__ mkey, float* __restrict__ ssum,
                              float* __restrict__ agg3, int E) {
  int e = blockIdx.x * blockDim.x + threadIdx.x;
  if (e >= E) return;
  int s = src[e], d = dst[e];
  float w = __expf(scores[e] - key2f(mkey[d]));
  atomicAdd(ssum + d, w);
  const float* v = O3 + (size_t)s * 64 + 32;   // v at col 32
  atomicAdd(agg3 + 3 * (size_t)d + 0, v[0] * w);
  atomicAdd(agg3 + 3 * (size_t)d + 1, v[1] * w);
  atomicAdd(agg3 + 3 * (size_t)d + 2, v[2] * w);
}

__global__ void k_final_out(const float* __restrict__ O3, const float* __restrict__ agg3,
                            const float* __restrict__ ssum, const float* __restrict__ noise,
                            float* __restrict__ out, int N) {
  int i = blockIdx.x * blockDim.x + threadIdx.x;
  if (i >= N * 3) return;
  int n = i / 3, c = i % 3;
  float drift = agg3[i] / (ssum[n] + 1e-16f) + O3[(size_t)n * 64 + 48 + c];  // skip at col 48
  out[i] = drift + noise[i] * (0.01f / 0.1f);  // noise*sigma/SQRT_D
}

// ---------------- launcher ----------------
extern "C" void kernel_launch(void* const* d_in, const int* in_sizes, int n_in,
                              void* d_out, int out_size, void* d_ws, size_t ws_size,
                              hipStream_t stream) {
  const float* x     = (const float*)d_in[0];
  const int*   eidx  = (const int*)d_in[1];
  const float* noise = (const float*)d_in[2];
  const int N = in_sizes[0] / 64;
  const int E = in_sizes[1] / 2;
  const int* src = eidx;       // edge_index[0]
  const int* dst = eidx + E;   // edge_index[1]

  const float *W1[4], *b1[4], *W2[4], *b2[4], *W3[4], *b3[4];
  for (int j = 0; j < 4; ++j) {
    W1[j] = (const float*)d_in[3 + 2 * j];       b1[j] = (const float*)d_in[4 + 2 * j];
    W2[j] = (const float*)d_in[11 + 2 * j];      b2[j] = (const float*)d_in[12 + 2 * j];
    W3[j] = (const float*)d_in[19 + 2 * j];      b3[j] = (const float*)d_in[20 + 2 * j];
  }

  // ---- carve workspace (256B aligned slices) ----
  char* p = (char*)d_ws;
  auto take = [&](size_t bytes) -> void* {
    p = (char*)(((uintptr_t)p + 255) & ~(uintptr_t)255);
    void* r = (void*)p;
    p += bytes;
    return r;
  };
  _Float16* xh   = (_Float16*)take((size_t)N * 64 * sizeof(_Float16)); // GEMM input (x, then h)
  _Float16* Wf1  = (_Float16*)take(16 * 1024 * sizeof(_Float16));      // frag-packed [64x256]
  _Float16* Wf2  = (_Float16*)take(16 * 1024 * sizeof(_Float16));
  _Float16* Wf3  = (_Float16*)take(4 * 1024 * sizeof(_Float16));       // frag-packed [64x64]
  float*    bf1  = (float*)take(256 * sizeof(float));
  float*    bf2  = (float*)take(256 * sizeof(float));
  float*    bf3  = (float*)take(64 * sizeof(float));
  float*    qkvs = (float*)take((size_t)N * 256 * sizeof(float));      // also holds O3 [N x 64]
  float*    h    = (float*)take((size_t)N * 64 * sizeof(float));
  float*    agg  = (float*)take((size_t)N * 64 * sizeof(float));       // reused as agg3 [N x 3]
  float*    scr  = (float*)take((size_t)E * sizeof(float));
  unsigned* mkey = (unsigned*)take((size_t)N * sizeof(unsigned));
  float*    ssum = (float*)take((size_t)N * sizeof(float));

  const int T = 256;
  const int gN64  = (N * 64 + T - 1) / T;
  const int gEw   = (E + WPB - 1) / WPB;    // one wave per edge
  const int gEt   = (E + T - 1) / T;        // one thread per edge
  const int tiles_m = N / 16;               // 100000/16 = 6250 exactly

  // ---- prep ----
  k_f32_to_f16<<<gN64, T, 0, stream>>>(x, xh, N * 64);
  k_prep_frag<<<(16 * 1024 + T - 1) / T, T, 0, stream>>>(W1[0], W1[1], W1[2], W1[3], Wf1, 256, 64);
  k_prep_frag<<<(16 * 1024 + T - 1) / T, T, 0, stream>>>(W2[0], W2[1], W2[2], W2[3], Wf2, 256, 64);
  k_prep_frag<<<(4 * 1024 + T - 1) / T, T, 0, stream>>>(W3[0], W3[1], W3[2], W3[3], Wf3, 64, 3);
  k_prep_bias<<<1, 256, 0, stream>>>(b1[0], b1[1], b1[2], b1[3], bf1, 256, 64);
  k_prep_bias<<<1, 256, 0, stream>>>(b2[0], b2[1], b2[2], b2[3], bf2, 256, 64);
  k_prep_bias<<<1, 64, 0, stream>>>(b3[0], b3[1], b3[2], b3[3], bf3, 64, 3);

  const float scale64 = 0.125f;            // 1/sqrt(64)
  const float scale3  = 0.57735026919f;    // 1/sqrt(3)

  // ---- layer 1 ----
  k_gemm_wmma<<<(tiles_m * 4 + WPB - 1) / WPB, T, 0, stream>>>(xh, Wf1, bf1, qkvs, tiles_m, 256);
  k_init_seg<<<gN64, T, 0, stream>>>(mkey, ssum, agg, N, N * 64);
  k_edge_scores64<<<gEw, T, 0, stream>>>(qkvs, src, dst, scr, mkey, E, scale64);
  k_edge_accum64<<<gEw, T, 0, stream>>>(qkvs, scr, src, dst, mkey, ssum, agg, E);
  k_finalize64<<<gN64, T, 0, stream>>>(qkvs, agg, ssum, h, xh, N, 1);

  // ---- layer 2 (residual) ----
  k_gemm_wmma<<<(tiles_m * 4 + WPB - 1) / WPB, T, 0, stream>>>(xh, Wf2, bf2, qkvs, tiles_m, 256);
  k_init_seg<<<gN64, T, 0, stream>>>(mkey, ssum, agg, N, N * 64);
  k_edge_scores64<<<gEw, T, 0, stream>>>(qkvs, src, dst, scr, mkey, E, scale64);
  k_edge_accum64<<<gEw, T, 0, stream>>>(qkvs, scr, src, dst, mkey, ssum, agg, E);
  k_finalize64<<<gN64, T, 0, stream>>>(qkvs, agg, ssum, h, xh, N, 2);

  // ---- layer 3 (D_OUT=3, padded fused GEMM to [N x 64]) ----
  k_gemm_wmma<<<(tiles_m * 1 + WPB - 1) / WPB, T, 0, stream>>>(xh, Wf3, bf3, qkvs, tiles_m, 64);
  k_init_seg<<<gN64, T, 0, stream>>>(mkey, ssum, agg, N, N * 3);
  k_edge3_scores<<<gEt, T, 0, stream>>>(qkvs, src, dst, scr, mkey, E, scale3);
  k_edge3_accum<<<gEt, T, 0, stream>>>(qkvs, scr, src, dst, mkey, ssum, agg, E);
  k_final_out<<<(N * 3 + T - 1) / T, T, 0, stream>>>(qkvs, agg, ssum, noise, (float*)d_out, N);
}
